// MQOTLayer_64931315581176
// MI455X (gfx1250) — matrix operand order repaired and checked by reading, not compiled, not run on hardware
//
#include <hip/hip_runtime.h>
#include <hip/hip_bf16.h>
#include <math.h>

// Problem constants (fixed by the reference's setup_inputs)
#define BB   16
#define TA   2048
#define TV   1024
#define DD   512
#define EPS  0.15f
#define NIT  20

typedef _Float16 __attribute__((ext_vector_type(4)))  v4h;
typedef _Float16 __attribute__((ext_vector_type(8)))  v8h;
typedef _Float16 __attribute__((ext_vector_type(16))) v16h;
typedef float    __attribute__((ext_vector_type(8)))  v8f;

// ---------------------------------------------------------------------------
// 1) L2-normalize rows of a [nrows, dim] f32 matrix, emit f16. One wave/row.
// ---------------------------------------------------------------------------
__global__ __launch_bounds__(256)
void normalize_rows_kernel(const float* __restrict__ x, _Float16* __restrict__ y,
                           int nrows, int dim)
{
    const int wid  = blockIdx.x * (blockDim.x >> 5) + (threadIdx.x >> 5);
    const int lane = threadIdx.x & 31;
    if (wid >= nrows) return;                    // uniform per wave

    const float* p = x + (size_t)wid * dim;
    float ss = 0.f;
    for (int k = lane * 4; k < dim; k += 128) {
        float4 f = *(const float4*)(p + k);
        ss += f.x * f.x + f.y * f.y + f.z * f.z + f.w * f.w;
    }
#pragma unroll
    for (int off = 16; off > 0; off >>= 1)
        ss += __shfl_xor(ss, off, 32);

    const float scale = 1.0f / fmaxf(sqrtf(ss), 1e-12f);

    _Float16* q = y + (size_t)wid * dim;
    for (int k = lane * 4; k < dim; k += 128) {
        float4 f = *(const float4*)(p + k);
        v4h h;
        h[0] = (_Float16)(f.x * scale);
        h[1] = (_Float16)(f.y * scale);
        h[2] = (_Float16)(f.z * scale);
        h[3] = (_Float16)(f.w * scale);
        *(v4h*)(q + k) = h;
    }
}

// ---------------------------------------------------------------------------
// 2) WMMA GEMM:  sim = a · v^T  (per batch), fused into K = -cost/eps.
//    Block = 256 threads (8 waves). Block tile 128(M) x 128(N).
//    Wave (wm in 0..1, wn in 0..3) owns a 64x32 region = 4x2 WMMA tiles.
// ---------------------------------------------------------------------------
__global__ __launch_bounds__(256)
void gemm_cost_kernel(const _Float16* __restrict__ a_h,
                      const _Float16* __restrict__ v_h,
                      const float*    __restrict__ quality,
                      const float*    __restrict__ lt_p,
                      const float*    __restrict__ lq_p,
                      float*          __restrict__ Kmat)
{
    const int b    = blockIdx.z;
    const int row0 = blockIdx.y * 128;
    const int col0 = blockIdx.x * 128;
    const int wave = threadIdx.x >> 5;
    const int lane = threadIdx.x & 31;
    const int wm   = wave >> 2;          // 0..1
    const int wn   = wave & 3;           // 0..3
    const int wrow = row0 + wm * 64;
    const int wcol = col0 + wn * 32;
    const int lh   = lane >> 4;          // half-wave id
    const int ll   = lane & 15;

    const _Float16* Abase = a_h + (size_t)b * TA * DD;
    const _Float16* Bbase = v_h + (size_t)b * TV * DD;

    v8f acc[4][2] = {};

    for (int k0 = 0; k0 < DD; k0 += 32) {
        // A fragments: 16x32 f16. Lane lh=0 rows hold K k0+0..7 and k0+16..23,
        // lane lh=1 rows hold K k0+8..15 and k0+24..31 (ISA 7.12.2 layout).
        v16h afrag[4];
#pragma unroll
        for (int mt = 0; mt < 4; ++mt) {
            const _Float16* p = Abase + (size_t)(wrow + mt * 16 + ll) * DD
                                      + k0 + 8 * lh;
            union { v16h v; v8h h[2]; } t;
            t.h[0] = *(const v8h*)(p);        // K = base+0..7
            t.h[1] = *(const v8h*)(p + 16);   // K = base+16..23
            afrag[mt] = t.v;
        }
        // B fragments: 32x16 f16. Lane holds column n = ll; elements 0..7 map to
        // K = k0 + 8*lh + e, elements 8..15 to K = k0 + 16 + 8*lh + (e-8)
        // (same order as A), read from row n of v (= column of v^T).
        v16h bfrag[2];
#pragma unroll
        for (int nt = 0; nt < 2; ++nt) {
            const _Float16* p = Bbase + (size_t)(wcol + nt * 16 + ll) * DD
                                      + k0 + 8 * lh;
            union { v16h v; v8h h[2]; } t;
            t.h[0] = *(const v8h*)(p);
            t.h[1] = *(const v8h*)(p + 16);
            bfrag[nt] = t.v;
        }
#pragma unroll
        for (int mt = 0; mt < 4; ++mt)
#pragma unroll
            for (int nt = 0; nt < 2; ++nt)
                acc[mt][nt] = __builtin_amdgcn_wmma_f32_16x16x32_f16(
                    false, afrag[mt], false, bfrag[nt],
                    (short)0, acc[mt][nt], false, false);
    }

    const float lt = lt_p[0];
    const float lq = lq_p[0];
    const float inv_eps = 1.0f / EPS;

    // C/D layout: acc[..][..][r] holds M = r + 8*lh, N = ll within the tile.
#pragma unroll
    for (int nt = 0; nt < 2; ++nt) {
        const int   j  = wcol + nt * 16 + ll;
        const float qp = lq * (1.0f - quality[(size_t)b * TV + j]);
        const float jn = (float)j / (float)TV;
#pragma unroll
        for (int mt = 0; mt < 4; ++mt) {
#pragma unroll
            for (int r = 0; r < 8; ++r) {
                const int   i    = wrow + mt * 16 + r + 8 * lh;
                const float sim  = acc[mt][nt][r];
                const float cost = (1.0f - sim)
                                 + lt * fabsf((float)i / (float)TA - jn)
                                 + qp;
                Kmat[((size_t)b * TA + i) * TV + j] = -cost * inv_eps;
            }
        }
    }
}

// ---------------------------------------------------------------------------
// 3) Row update: u[b,i] = -logsumexp_j(K[b,i,j] + w[b,j]).  One wave per row;
//    the 1024-float row lives in 32 regs/lane (8 x float4).
// ---------------------------------------------------------------------------
__global__ __launch_bounds__(256)
void row_lse_kernel(const float* __restrict__ Kmat,
                    const float* __restrict__ w,
                    float*       __restrict__ u)
{
    const int wid  = blockIdx.x * (blockDim.x >> 5) + (threadIdx.x >> 5);
    const int lane = threadIdx.x & 31;
    const int b = wid / TA;
    const int i = wid % TA;

    const float* krow = Kmat + ((size_t)b * TA + i) * TV;
    const float* wrow = w + (size_t)b * TV;

    float4 xs[8];
    float  m = -INFINITY;
#pragma unroll
    for (int t = 0; t < 8; ++t) {
        const int j = t * 128 + lane * 4;
        float4 k4 = *(const float4*)(krow + j);
        float4 w4 = *(const float4*)(wrow + j);
        float4 x;
        x.x = k4.x + w4.x; x.y = k4.y + w4.y;
        x.z = k4.z + w4.z; x.w = k4.w + w4.w;
        xs[t] = x;
        m = fmaxf(m, fmaxf(fmaxf(x.x, x.y), fmaxf(x.z, x.w)));
    }
#pragma unroll
    for (int off = 16; off > 0; off >>= 1)
        m = fmaxf(m, __shfl_xor(m, off, 32));

    float s = 0.f;
#pragma unroll
    for (int t = 0; t < 8; ++t) {
        s += __expf(xs[t].x - m) + __expf(xs[t].y - m)
           + __expf(xs[t].z - m) + __expf(xs[t].w - m);
    }
#pragma unroll
    for (int off = 16; off > 0; off >>= 1)
        s += __shfl_xor(s, off, 32);

    if (lane == 0)
        u[(size_t)b * TA + i] = -(m + __logf(s));
}

// ---------------------------------------------------------------------------
// 4) Col update: w[b,j] = -logsumexp_i(K[b,i,j] + u[b,i]).
//    Block = 256 threads handling 32 columns; warp p walks rows p, p+8, ...
//    with coalesced 128B row-segments and a streaming (online) logsumexp,
//    then an LDS combine across the 8 warps.
// ---------------------------------------------------------------------------
__global__ __launch_bounds__(256)
void col_lse_kernel(const float* __restrict__ Kmat,
                    const float* __restrict__ u,
                    float*       __restrict__ w)
{
    const int b    = blockIdx.y;
    const int j    = blockIdx.x * 32 + (threadIdx.x & 31);
    const int warp = threadIdx.x >> 5;

    const float* kb = Kmat + (size_t)b * TA * TV;
    const float* ub = u + (size_t)b * TA;

    float m = -INFINITY, s = 0.f;
    for (int i = warp; i < TA; i += 8) {
        const float x = kb[(size_t)i * TV + j] + ub[i];
        if (x > m) { s = s * __expf(m - x) + 1.0f; m = x; }
        else       { s += __expf(x - m); }
    }

    __shared__ float sm[8][32];
    __shared__ float ssum[8][32];
    sm[warp][threadIdx.x & 31]   = m;
    ssum[warp][threadIdx.x & 31] = s;
    __syncthreads();

    if (threadIdx.x < 32) {
        float M = sm[0][threadIdx.x], S = ssum[0][threadIdx.x];
#pragma unroll
        for (int p = 1; p < 8; ++p) {
            const float m2 = sm[p][threadIdx.x], s2 = ssum[p][threadIdx.x];
            const float Mn = fmaxf(M, m2);
            S = S * __expf(M - Mn) + s2 * __expf(m2 - Mn);
            M = Mn;
        }
        w[(size_t)b * TV + j] = -(M + __logf(S));
    }
}

// ---------------------------------------------------------------------------
// 5) transport = exp(K + u_i + w_j), in place on K (== d_out). float4/thread.
// ---------------------------------------------------------------------------
__global__ __launch_bounds__(256)
void transport_kernel(float* __restrict__ Kmat,
                      const float* __restrict__ u,
                      const float* __restrict__ w)
{
    const size_t lin = ((size_t)blockIdx.x * blockDim.x + threadIdx.x) * 4;
    const int j = (int)(lin % TV);
    const int i = (int)((lin / TV) % TA);
    const int b = (int)(lin / ((size_t)TA * TV));

    const float uv = u[(size_t)b * TA + i];
    float4 k4 = *(const float4*)(Kmat + lin);
    const float* wj = w + (size_t)b * TV + j;
    float4 o;
    o.x = __expf(k4.x + uv + wj[0]);
    o.y = __expf(k4.y + uv + wj[1]);
    o.z = __expf(k4.z + uv + wj[2]);
    o.w = __expf(k4.w + uv + wj[3]);
    *(float4*)(Kmat + lin) = o;
}

__global__ __launch_bounds__(256)
void zero_kernel(float* __restrict__ p, int n)
{
    const int idx = blockIdx.x * blockDim.x + threadIdx.x;
    if (idx < n) p[idx] = 0.f;
}

// ---------------------------------------------------------------------------
// Launcher
// ---------------------------------------------------------------------------
extern "C" void kernel_launch(void* const* d_in, const int* in_sizes, int n_in,
                              void* d_out, int out_size, void* d_ws, size_t ws_size,
                              hipStream_t stream)
{
    const float* audio   = (const float*)d_in[0];   // [B,Ta,D]
    const float* video   = (const float*)d_in[1];   // [B,Tv,D]
    const float* quality = (const float*)d_in[2];   // [B,Tv]
    const float* lt_p    = (const float*)d_in[3];   // scalar
    const float* lq_p    = (const float*)d_in[4];   // scalar

    float* Kmat = (float*)d_out;                    // K matrix lives in d_out

    // Workspace carve-up (aligned generously)
    char* ws = (char*)d_ws;
    _Float16* a_h = (_Float16*)(ws);                               // 32 MB
    _Float16* v_h = (_Float16*)(ws + (size_t)BB * TA * DD * 2);    // 16 MB
    float*    u   = (float*)(ws + (size_t)BB * TA * DD * 2
                                 + (size_t)BB * TV * DD * 2);      // 128 KB
    float*    w   = u + (size_t)BB * TA;                           // 64 KB

    // 1) normalize + f16 convert
    {
        const int rows_a = BB * TA;                 // 32768
        const int rows_v = BB * TV;                 // 16384
        normalize_rows_kernel<<<rows_a / 8, 256, 0, stream>>>(audio, a_h, rows_a, DD);
        normalize_rows_kernel<<<rows_v / 8, 256, 0, stream>>>(video, v_h, rows_v, DD);
    }

    // 2) WMMA GEMM fused with cost -> K
    {
        dim3 grid(TV / 128, TA / 128, BB);          // (8,16,16)
        gemm_cost_kernel<<<grid, 256, 0, stream>>>(a_h, v_h, quality, lt_p, lq_p, Kmat);
    }

    // 3) Sinkhorn iterations (K stays L2-resident: 134 MB < 192 MB L2)
    zero_kernel<<<(BB * TV + 255) / 256, 256, 0, stream>>>(w, BB * TV);
    for (int it = 0; it < NIT; ++it) {
        row_lse_kernel<<<(BB * TA) / 8, 256, 0, stream>>>(Kmat, w, u);
        col_lse_kernel<<<dim3(TV / 32, BB), 256, 0, stream>>>(Kmat, u, w);
    }

    // 4) transport = exp(K + u + w), in place
    {
        const size_t total = (size_t)BB * TA * TV;  // 33.5M
        transport_kernel<<<(unsigned)(total / 4 / 256), 256, 0, stream>>>(Kmat, u, w);
    }
}
